// StyleLinearLayer_9766755631183
// MI455X (gfx1250) — matrix operand order repaired and checked by reading, not compiled, not run on hardware
//
#include <hip/hip_runtime.h>

// ---------------------------------------------------------------------------
// StyleLinearLayer fused kernel for MI455X (gfx1250, wave32, WMMA).
//   out = leaky_relu( ((X @ W^T + b) * w) @ W^T * dcoefs + noise + b )
// bf16x3 split-precision GEMMs on v_wmma_f32_16x16x32_bf16; both GEMMs fused
// per 16-row strip with the intermediate kept in LDS.
// v2: 2-tile register blocking per wave + unroll-1 K-loop to kill the
//     scratch spills seen in v1's unrolled inner loop.
// ---------------------------------------------------------------------------

#define DDIM    1024
#define MBLK    16
#define THREADS 256
#define NWAVES  8      // 256 threads / wave32
#define NTILES  (DDIM / 16)   // 64 column tiles
#define NPAIRS  (NTILES / 2)  // 32 tile pairs

typedef __attribute__((ext_vector_type(16))) __bf16         v16bf;
typedef __attribute__((ext_vector_type(8)))  float          v8f;
typedef __attribute__((ext_vector_type(16))) unsigned short u16x16;
typedef __attribute__((ext_vector_type(8)))  unsigned short u16x8;

union Frag {
    v16bf  v;
    u16x16 u;
    u16x8  h[2];
};

// float -> bf16 with round-to-nearest-even (bit ops; no __bf16 arithmetic).
static __device__ __forceinline__ unsigned short f2bf(float f) {
    unsigned int u = __float_as_uint(f);
    u += 0x7FFFu + ((u >> 16) & 1u);
    return (unsigned short)(u >> 16);
}
static __device__ __forceinline__ float bf2f(unsigned short h) {
    return __uint_as_float(((unsigned int)h) << 16);
}

// ---------------------------------------------------------------------------
// Prep kernel: one block per output row o of W.
//  - split W[o,:] into bf16 hi/lo           -> Whi, Wlo      (GEMM1 weights)
//  - split (W[o,:] * w[:]) into bf16 hi/lo  -> Mhi, Mlo      (GEMM2 weights)
//  - dcoefs[o] = rsqrt(sum_j (W[o,j]*w[j])^2 + 1e-8)   (fp32 reduction)
//  - c2[o]     = noise_raw[o]*noise_strength + b[o]    (epilogue constant)
// ---------------------------------------------------------------------------
__launch_bounds__(THREADS)
__global__ void style_prep_kernel(const float* __restrict__ W,
                                  const float* __restrict__ w,
                                  const float* __restrict__ b,
                                  const float* __restrict__ ns,
                                  const float* __restrict__ nraw,
                                  unsigned short* __restrict__ Whi,
                                  unsigned short* __restrict__ Wlo,
                                  unsigned short* __restrict__ Mhi,
                                  unsigned short* __restrict__ Mlo,
                                  float* __restrict__ dcoefs,
                                  float* __restrict__ c2) {
    const int o = blockIdx.x;
    const int t = threadIdx.x;
    const float* row = W + (size_t)o * DDIM;

    float sum = 0.0f;
    for (int j = t; j < DDIM; j += THREADS) {
        float wv = row[j];
        float wm = wv * w[j];
        unsigned short h0 = f2bf(wv);
        Whi[(size_t)o * DDIM + j] = h0;
        Wlo[(size_t)o * DDIM + j] = f2bf(wv - bf2f(h0));
        unsigned short h1 = f2bf(wm);
        Mhi[(size_t)o * DDIM + j] = h1;
        Mlo[(size_t)o * DDIM + j] = f2bf(wm - bf2f(h1));
        sum += wm * wm;
    }

    __shared__ float red[THREADS];
    red[t] = sum;
    __syncthreads();
    for (int s = THREADS / 2; s > 0; s >>= 1) {
        if (t < s) red[t] += red[t + s];
        __syncthreads();
    }
    if (t == 0) {
        dcoefs[o] = rsqrtf(red[0] + 1e-8f);
        c2[o] = nraw[o] * ns[0] + b[o];
    }
}

// ---------------------------------------------------------------------------
// Fused double-GEMM kernel. One block = one 16-row strip of X.
// Phase 1: X1 = X @ W^T + b   -> LDS (split bf16 hi/lo)
// Phase 2: out = leaky_relu( (X1 @ Wmod^T) * dcoefs + c2 )
// Each wave owns 4 PAIRS of adjacent 16-col tiles per phase; A fragments are
// loaded once per K-step and shared across the two accumulator chains.
// bf16x3: acc += aHi*bHi + aHi*bLo + aLo*bHi  (f32 accumulation)
// ---------------------------------------------------------------------------
__launch_bounds__(THREADS)
__global__ void style_fused_kernel(const float* __restrict__ x,
                                   const float* __restrict__ b,
                                   const unsigned short* __restrict__ Whi,
                                   const unsigned short* __restrict__ Wlo,
                                   const unsigned short* __restrict__ Mhi,
                                   const unsigned short* __restrict__ Mlo,
                                   const float* __restrict__ dcoefs,
                                   const float* __restrict__ c2,
                                   float* __restrict__ out) {
    __shared__ __attribute__((aligned(32))) unsigned short sXhi[MBLK * DDIM];
    __shared__ __attribute__((aligned(32))) unsigned short sXlo[MBLK * DDIM];
    __shared__ __attribute__((aligned(32))) unsigned short sYhi[MBLK * DDIM];
    __shared__ __attribute__((aligned(32))) unsigned short sYlo[MBLK * DDIM];

    const int tid     = threadIdx.x;
    const int lane    = tid & 31;
    const int wave    = tid >> 5;
    const int ln15    = lane & 15;
    const int hiHalf  = lane >> 4;                // 0: lanes 0-15, 1: lanes 16-31
    const size_t rowBase = (size_t)blockIdx.x * MBLK;
    const int arow = ln15 * DDIM;

    // ---- Stage the 16x1024 fp32 strip of X into LDS as split bf16 ----
    for (int i = tid; i < MBLK * DDIM; i += THREADS) {
        float v = x[rowBase * DDIM + (size_t)i];
        unsigned short h = f2bf(v);
        sXhi[i] = h;
        sXlo[i] = f2bf(v - bf2f(h));
    }
    __syncthreads();

    // ---- Phase 1: X1 strip = X @ W^T + b ----
    for (int tp = wave; tp < NPAIRS; tp += NWAVES) {
        const int obase = tp * 32;                // two adjacent 16-col tiles
        const int o0    = obase + ln15;           // lane's column, tile 0
        const int o1    = o0 + 16;                // lane's column, tile 1
        v8f acc0 = {};
        v8f acc1 = {};
        const unsigned short* bHiRow = Whi + (size_t)o0 * DDIM;
        const unsigned short* bLoRow = Wlo + (size_t)o0 * DDIM;
        // tile-1 rows are at a constant +16*DDIM element offset (fits IOFFSET)

#pragma unroll 1
        for (int k0 = 0; k0 < DDIM; k0 += 32) {
            // A (16x32 bf16): lanes 0-15 hold K k0+[0..15]; lanes 16-31 +16
            const int kA = k0 + hiHalf * 16;
            Frag aHi, aLo;
            aHi.h[0] = *(const u16x8*)&sXhi[arow + kA];
            aHi.h[1] = *(const u16x8*)&sXhi[arow + kA + 8];
            aLo.h[0] = *(const u16x8*)&sXlo[arow + kA];
            aLo.h[1] = *(const u16x8*)&sXlo[arow + kA + 8];
            // B (32x16 bf16): lane holds 16 contiguous K values of its column
            const int kB = k0 + hiHalf * 16;
            __builtin_prefetch((const void*)(bHiRow + kB + 256), 0, 1);
            Frag bHi0, bLo0, bHi1, bLo1;
            bHi0.u = *(const u16x16*)&bHiRow[kB];
            bLo0.u = *(const u16x16*)&bLoRow[kB];
            bHi1.u = *(const u16x16*)&bHiRow[16 * DDIM + kB];
            bLo1.u = *(const u16x16*)&bLoRow[16 * DDIM + kB];

            // two independent accumulator chains, interleaved
            acc0 = __builtin_amdgcn_wmma_f32_16x16x32_bf16(false, aHi.v, false, bHi0.v,
                                                           (short)0, acc0, false, false);
            acc1 = __builtin_amdgcn_wmma_f32_16x16x32_bf16(false, aHi.v, false, bHi1.v,
                                                           (short)0, acc1, false, false);
            acc0 = __builtin_amdgcn_wmma_f32_16x16x32_bf16(false, aHi.v, false, bLo0.v,
                                                           (short)0, acc0, false, false);
            acc1 = __builtin_amdgcn_wmma_f32_16x16x32_bf16(false, aHi.v, false, bLo1.v,
                                                           (short)0, acc1, false, false);
            acc0 = __builtin_amdgcn_wmma_f32_16x16x32_bf16(false, aLo.v, false, bHi0.v,
                                                           (short)0, acc0, false, false);
            acc1 = __builtin_amdgcn_wmma_f32_16x16x32_bf16(false, aLo.v, false, bHi1.v,
                                                           (short)0, acc1, false, false);
        }

        // Epilogue: + bias, split-store into LDS X1 strip.
        // C/D layout: VGPR r -> M = r + 8*hiHalf, N = lane%16.
        const float bias0 = b[o0];
        const float bias1 = b[o1];
#pragma unroll
        for (int r = 0; r < 8; ++r) {
            const int m = r + hiHalf * 8;
            float v0 = acc0[r] + bias0;
            unsigned short h0 = f2bf(v0);
            sYhi[m * DDIM + o0] = h0;
            sYlo[m * DDIM + o0] = f2bf(v0 - bf2f(h0));
            float v1 = acc1[r] + bias1;
            unsigned short h1 = f2bf(v1);
            sYhi[m * DDIM + o1] = h1;
            sYlo[m * DDIM + o1] = f2bf(v1 - bf2f(h1));
        }
    }
    __syncthreads();

    // ---- Phase 2: out = leaky_relu( (X1 @ Wmod^T) * dcoefs + c2 ) ----
    for (int tp = wave; tp < NPAIRS; tp += NWAVES) {
        const int nbase = tp * 32;
        const int o0    = nbase + ln15;
        const int o1    = o0 + 16;
        v8f acc0 = {};
        v8f acc1 = {};
        const unsigned short* bHiRow = Mhi + (size_t)o0 * DDIM;
        const unsigned short* bLoRow = Mlo + (size_t)o0 * DDIM;

#pragma unroll 1
        for (int k0 = 0; k0 < DDIM; k0 += 32) {
            const int kA = k0 + hiHalf * 16;
            Frag aHi, aLo;
            aHi.h[0] = *(const u16x8*)&sYhi[arow + kA];
            aHi.h[1] = *(const u16x8*)&sYhi[arow + kA + 8];
            aLo.h[0] = *(const u16x8*)&sYlo[arow + kA];
            aLo.h[1] = *(const u16x8*)&sYlo[arow + kA + 8];
            const int kB = k0 + hiHalf * 16;
            __builtin_prefetch((const void*)(bHiRow + kB + 256), 0, 1);
            Frag bHi0, bLo0, bHi1, bLo1;
            bHi0.u = *(const u16x16*)&bHiRow[kB];
            bLo0.u = *(const u16x16*)&bLoRow[kB];
            bHi1.u = *(const u16x16*)&bHiRow[16 * DDIM + kB];
            bLo1.u = *(const u16x16*)&bLoRow[16 * DDIM + kB];

            acc0 = __builtin_amdgcn_wmma_f32_16x16x32_bf16(false, aHi.v, false, bHi0.v,
                                                           (short)0, acc0, false, false);
            acc1 = __builtin_amdgcn_wmma_f32_16x16x32_bf16(false, aHi.v, false, bHi1.v,
                                                           (short)0, acc1, false, false);
            acc0 = __builtin_amdgcn_wmma_f32_16x16x32_bf16(false, aHi.v, false, bLo0.v,
                                                           (short)0, acc0, false, false);
            acc1 = __builtin_amdgcn_wmma_f32_16x16x32_bf16(false, aHi.v, false, bLo1.v,
                                                           (short)0, acc1, false, false);
            acc0 = __builtin_amdgcn_wmma_f32_16x16x32_bf16(false, aLo.v, false, bHi0.v,
                                                           (short)0, acc0, false, false);
            acc1 = __builtin_amdgcn_wmma_f32_16x16x32_bf16(false, aLo.v, false, bHi1.v,
                                                           (short)0, acc1, false, false);
        }

        const float dc0 = dcoefs[o0];
        const float cc0 = c2[o0];
        const float dc1 = dcoefs[o1];
        const float cc1 = c2[o1];
#pragma unroll
        for (int r = 0; r < 8; ++r) {
            const int m = r + hiHalf * 8;
            float v0 = acc0[r] * dc0 + cc0;
            v0 = fmaxf(v0, 0.01f * v0);           // leaky_relu, slope 0.01
            out[(rowBase + (size_t)m) * DDIM + o0] = v0;
            float v1 = acc1[r] * dc1 + cc1;
            v1 = fmaxf(v1, 0.01f * v1);
            out[(rowBase + (size_t)m) * DDIM + o1] = v1;
        }
    }
}

// ---------------------------------------------------------------------------
// Host launcher. Inputs (setup_inputs order):
//   0: x [N,D] f32   1: w [D] f32   2: W [D,D] f32   3: b [D] f32
//   4: noise_strength [1] f32       5: noise_raw [1,D] f32
// Workspace: Whi|Wlo|Mhi|Mlo (2 MB each, bf16) + dcoefs + c2 (~8.4 MB total).
// ---------------------------------------------------------------------------
extern "C" void kernel_launch(void* const* d_in, const int* in_sizes, int n_in,
                              void* d_out, int out_size, void* d_ws, size_t ws_size,
                              hipStream_t stream) {
    (void)n_in; (void)out_size; (void)ws_size;
    const float* x  = (const float*)d_in[0];
    const float* w  = (const float*)d_in[1];
    const float* W  = (const float*)d_in[2];
    const float* b  = (const float*)d_in[3];
    const float* ns = (const float*)d_in[4];
    const float* nr = (const float*)d_in[5];
    float* out = (float*)d_out;

    char* ws = (char*)d_ws;
    const size_t wbytes = (size_t)DDIM * DDIM * sizeof(unsigned short); // 2 MB
    unsigned short* Whi = (unsigned short*)(ws + 0 * wbytes);
    unsigned short* Wlo = (unsigned short*)(ws + 1 * wbytes);
    unsigned short* Mhi = (unsigned short*)(ws + 2 * wbytes);
    unsigned short* Mlo = (unsigned short*)(ws + 3 * wbytes);
    float* dcoefs = (float*)(ws + 4 * wbytes);
    float* c2     = (float*)(ws + 4 * wbytes + DDIM * sizeof(float));

    const int nRows = in_sizes[0] / DDIM;   // N = 65536

    style_prep_kernel<<<DDIM, THREADS, 0, stream>>>(W, w, b, ns, nr,
                                                    Whi, Wlo, Mhi, Mlo, dcoefs, c2);
    style_fused_kernel<<<nRows / MBLK, THREADS, 0, stream>>>(x, b, Whi, Wlo, Mhi, Mlo,
                                                             dcoefs, c2, out);
}
